// Transformer_21698174779983
// MI455X (gfx1250) — matrix operand order repaired and checked by reading, not compiled, not run on hardware
//
#include <hip/hip_runtime.h>

// ---------------- model dims ----------------
#define DIM   1024
#define ADIM  1024
#define NHEAD 8
#define SEQL  1024
#define HTOT  (NHEAD * ADIM)   // 8192

// ---------------- WMMA types ----------------
typedef __attribute__((ext_vector_type(16))) __bf16 v16bf;
typedef __attribute__((ext_vector_type(8)))  __bf16 v8bf;
typedef __attribute__((ext_vector_type(4)))  __bf16 v4bf;
typedef __attribute__((ext_vector_type(8)))  float  v8f;
typedef __attribute__((ext_vector_type(4)))  int    v4i;

// ---------------- GEMM tiling ----------------
#define BM 128
#define BN 128
#define BK 64
#define LDSR 72            // padded LDS row stride (bf16): 144 B -> 16B-aligned, conflict-free b128
#define TILE_B (BM * LDSR * 2)   // 18432 bytes per tile buffer

// low 32 bits of a generic pointer to __shared__ == LDS byte offset
__device__ __forceinline__ uint32_t lds_off(const void* p) {
    return (uint32_t)(uint64_t)(uintptr_t)p;
}

// raw LDS b128 load: volatile asm keeps all fragment loads clustered in issue order
__device__ __forceinline__ v4i ds_ld128(uint32_t addr) {
    v4i r;
    asm volatile("ds_load_b128 %0, %1" : "=v"(r) : "v"(addr) : "memory");
    return r;
}

__device__ __forceinline__ v16bf mk16(v4i lo4, v4i hi4) {
    const v8bf lo = __builtin_bit_cast(v8bf, lo4);
    const v8bf hi = __builtin_bit_cast(v8bf, hi4);
    v16bf r;
    #pragma unroll
    for (int e = 0; e < 8; ++e) { r[e] = lo[e]; r[8 + e] = hi[e]; }
    return r;
}

// C = A(M x K) * Bt(N x K)^T + bias.  A, Bt bf16 (pre-converted; Bt = W transposed), C fp32.
// grid = (N/BN, M/BM), block = 256 (8 wave32).
// Double-buffered LDS tiles streamed with CDNA5 async-DMA (stage i+1 DMA overlaps stage i math).
// Fragment loads are hand-issued asm: 18 clustered ds_load_b128, ONE s_wait_dscnt 0 (which ties
// every fragment register as an operand so WMMAs cannot be hoisted above it and the registers
// cannot be coalesced), then 8 back-to-back WMMAs into the XDL pipe.
__global__ __launch_bounds__(256, 2) void gemm_bias_bf16wmma(
    const __bf16* __restrict__ A, const __bf16* __restrict__ Bt,
    const float* __restrict__ bias, float* __restrict__ C,
    int K, int N)
{
    __shared__ __align__(16) __bf16 sA[2][BM * LDSR];  // [m][k], 2 stages
    __shared__ __align__(16) __bf16 sB[2][BN * LDSR];  // [n][k], 2 stages

    const int tid  = threadIdx.x;
    const int wv   = tid >> 5;
    const int lane = tid & 31;
    const int hl   = lane >> 4;
    const int l16  = lane & 15;
    const int bm = blockIdx.y, bn = blockIdx.x;

    v8f acc[8];
    #pragma unroll
    for (int i = 0; i < 8; ++i)
        #pragma unroll
        for (int e = 0; e < 8; ++e) acc[i][e] = 0.0f;

    const char* Agb = (const char*)(A  + (size_t)bm * BM * K);
    const char* Bgb = (const char*)(Bt + (size_t)bn * BN * K);
    const int rr = tid >> 3;           // 0..31 : row within a 32-row pass
    const int cB = (tid & 7) * 16;     // byte offset of 16-byte chunk within a 128-byte row
    const uint32_t sAoff = lds_off(&sA[0][0]);
    const uint32_t sBoff = lds_off(&sB[0][0]);

    // issue one full stage (A tile + B tile, 128x64 bf16 each) via async DMA: 8 ops/thread
    auto issue = [&](int kt, int b) {
        #pragma unroll
        for (int p = 0; p < 4; ++p) {
            const int r = p * 32 + rr;
            asm volatile("global_load_async_to_lds_b128 %0, %1, off"
                :: "v"(sAoff + (uint32_t)(b * TILE_B + r * (LDSR * 2)) + (uint32_t)cB),
                   "v"(Agb + ((size_t)r * K + kt) * 2 + cB)
                : "memory");
            asm volatile("global_load_async_to_lds_b128 %0, %1, off"
                :: "v"(sBoff + (uint32_t)(b * TILE_B + r * (LDSR * 2)) + (uint32_t)cB),
                   "v"(Bgb + ((size_t)r * K + kt) * 2 + cB)
                : "memory");
        }
    };

    issue(0, 0);
    int buf = 0;
    for (int kt = 0; kt < K; kt += BK, buf ^= 1) {
        if (kt + BK < K) {
            issue(kt + BK, buf ^ 1);                       // prefetch next stage
            asm volatile("s_wait_asynccnt 0x8" ::: "memory");  // older 8 ops (this stage) done
        } else {
            asm volatile("s_wait_asynccnt 0x0" ::: "memory");
        }
        __syncthreads();

        const uint32_t tAo = sAoff + (uint32_t)buf * TILE_B;
        const uint32_t tBo = sBoff + (uint32_t)buf * TILE_B;
        const int m = wv * 16 + l16;
        #pragma unroll
        for (int kk = 0; kk < BK; kk += 32) {
            // A fragment halves: K = kk + hl*16 + {0..15}
            const uint32_t aoff = tAo + (uint32_t)((m * LDSR + kk + hl * 16) * 2);
            v4i a0 = ds_ld128(aoff);
            v4i a1 = ds_ld128(aoff + 16);
            // 8 B fragments (16 halves): lane col n; K = kk + hl*16 + {0..15}
            v4i b[16];
            #pragma unroll
            for (int ns = 0; ns < 8; ++ns) {
                const uint32_t boff =
                    tBo + (uint32_t)((((ns * 16 + l16) * LDSR) + kk + hl * 16) * 2);
                b[2 * ns]     = ds_ld128(boff);
                b[2 * ns + 1] = ds_ld128(boff + 16);
            }
            // single wait, tying every fragment register across it
            asm volatile("s_wait_dscnt 0x0"
                : "+v"(a0), "+v"(a1),
                  "+v"(b[0]),  "+v"(b[1]),  "+v"(b[2]),  "+v"(b[3]),
                  "+v"(b[4]),  "+v"(b[5]),  "+v"(b[6]),  "+v"(b[7]),
                  "+v"(b[8]),  "+v"(b[9]),  "+v"(b[10]), "+v"(b[11]),
                  "+v"(b[12]), "+v"(b[13]), "+v"(b[14]), "+v"(b[15])
                :: "memory");
            const v16bf a = mk16(a0, a1);
            #pragma unroll
            for (int ns = 0; ns < 8; ++ns) {
                const v16bf bb = mk16(b[2 * ns], b[2 * ns + 1]);
                acc[ns] = __builtin_amdgcn_wmma_f32_16x16x32_bf16(
                    false, a, false, bb, (short)0, acc[ns], false, false);
            }
        }
        __syncthreads();
    }

    // epilogue: C layout M = v + 8*hl, N = l16
    #pragma unroll
    for (int ns = 0; ns < 8; ++ns) {
        const int n = bn * BN + ns * 16 + l16;
        const float bv = bias[n];
        #pragma unroll
        for (int v = 0; v < 8; ++v) {
            const int row = bm * BM + wv * 16 + hl * 8 + v;
            C[(size_t)row * N + n] = acc[ns][v] + bv;
        }
    }
}

// ---------------- fp32 -> bf16 (flat) ----------------
__global__ __launch_bounds__(256) void convert_bf16(
    const float* __restrict__ x, __bf16* __restrict__ y)
{
    const size_t i = ((size_t)blockIdx.x * 256 + threadIdx.x) * 4;
    const float4 f = *(const float4*)(x + i);
    v4bf o;
    o[0] = (__bf16)f.x; o[1] = (__bf16)f.y; o[2] = (__bf16)f.z; o[3] = (__bf16)f.w;
    *(v4bf*)(y + i) = o;
}

// ---------------- W (K x N fp32) -> Wt (N x K bf16) ----------------
__global__ __launch_bounds__(256) void transpose_convert_bf16(
    const float* __restrict__ W, __bf16* __restrict__ Wt, int K, int N)
{
    __shared__ float t[32][33];
    const int k0 = blockIdx.y * 32, n0 = blockIdx.x * 32;
    const int tid = threadIdx.x;
    {
        const int k = tid >> 3, c = (tid & 7) * 4;
        const float4 f = *(const float4*)(W + (size_t)(k0 + k) * N + n0 + c);
        t[k][c] = f.x; t[k][c + 1] = f.y; t[k][c + 2] = f.z; t[k][c + 3] = f.w;
    }
    __syncthreads();
    {
        const int n = tid >> 3, c = (tid & 7) * 4;
        v4bf o;
        #pragma unroll
        for (int j = 0; j < 4; ++j) o[j] = (__bf16)t[c + j][n];
        *(v4bf*)(Wt + (size_t)(n0 + n) * K + k0 + c) = o;
    }
}

// ---------------- reductions ----------------
__device__ __forceinline__ float warpSum(float v) {
    #pragma unroll
    for (int o = 16; o > 0; o >>= 1) v += __shfl_xor(v, o, 32);
    return v;
}
__device__ __forceinline__ float warpMax(float v) {
    #pragma unroll
    for (int o = 16; o > 0; o >>= 1) v = fmaxf(v, __shfl_xor(v, o, 32));
    return v;
}
__device__ __forceinline__ float blockSum(float v, float* red) {
    v = warpSum(v);
    if ((threadIdx.x & 31) == 0) red[threadIdx.x >> 5] = v;
    __syncthreads();
    float r = 0.f;
    #pragma unroll
    for (int i = 0; i < 8; ++i) r += red[i];
    __syncthreads();
    return r;
}
__device__ __forceinline__ float blockMax(float v, float* red) {
    v = warpMax(v);
    if ((threadIdx.x & 31) == 0) red[threadIdx.x >> 5] = v;
    __syncthreads();
    float r = -3.4e38f;
    #pragma unroll
    for (int i = 0; i < 8; ++i) r = fmaxf(r, red[i]);
    __syncthreads();
    return r;
}

// ---------------- embedding + positional encoding ----------------
__global__ __launch_bounds__(256) void embed_pe(
    const int* __restrict__ idx, const float* __restrict__ emb,
    float* __restrict__ out)
{
    const int l = blockIdx.x;
    const int d = blockIdx.y * 256 + threadIdx.x;
    const float ex    = (float)(d >> 1) * (1.0f / (float)DIM);
    const float denom = powf(10000.0f, ex);
    const float ang   = (float)l / denom;
    const float pe    = (d & 1) ? cosf(ang) : sinf(ang);
    out[(size_t)l * DIM + d] = emb[(size_t)idx[l] * DIM + d] + pe;
}

// ---------------- elementwise attention + LN + softmax + *V ----------------
__global__ __launch_bounds__(256) void attn_elem(
    const float* __restrict__ Q, const float* __restrict__ Km,
    const float* __restrict__ V, const float* __restrict__ g,
    const float* __restrict__ bt, float* __restrict__ O)
{
    const int h = blockIdx.x >> 10;
    const int i = blockIdx.x & (SEQL - 1);
    __shared__ float row[ADIM];
    __shared__ float red[8];
    const int tid = threadIdx.x;
    const size_t base = (size_t)i * HTOT + (size_t)h * ADIM;
    const float scale = 0.03125f;             // 1/sqrt(1024)

    float s = 0.f, s2 = 0.f;
    for (int j = tid; j < ADIM; j += 256) {
        const float a = Q[base + j] *
                        Km[(size_t)j * HTOT + (size_t)h * ADIM + i] * scale;
        row[j] = a; s += a; s2 += a * a;
    }
    const float S  = blockSum(s,  red);
    const float S2 = blockSum(s2, red);
    const float mean = S * (1.f / ADIM);
    const float inv  = rsqrtf(S2 * (1.f / ADIM) - mean * mean + 1e-5f);

    float mx = -3.4e38f;
    for (int j = tid; j < ADIM; j += 256) {
        const float a = (row[j] - mean) * inv * g[j] + bt[j];
        row[j] = a;
        mx = fmaxf(mx, a);
    }
    const float MX = blockMax(mx, red);

    float es = 0.f;
    for (int j = tid; j < ADIM; j += 256) {
        const float e = __expf(row[j] - MX);
        row[j] = e; es += e;
    }
    const float rs = 1.0f / blockSum(es, red);
    for (int j = tid; j < ADIM; j += 256)
        O[base + j] = row[j] * rs * V[base + j];
}

// ---------------- residual + layernorm ----------------
__global__ __launch_bounds__(256) void add_ln(
    const float* __restrict__ x, const float* __restrict__ res,
    const float* __restrict__ g, const float* __restrict__ b,
    float* __restrict__ out)
{
    const int i = blockIdx.x;
    __shared__ float row[DIM];
    __shared__ float red[8];
    const int tid = threadIdx.x;
    float s = 0.f, s2 = 0.f;
    for (int j = tid; j < DIM; j += 256) {
        const float a = x[(size_t)i * DIM + j] + res[(size_t)i * DIM + j];
        row[j] = a; s += a; s2 += a * a;
    }
    const float S  = blockSum(s,  red);
    const float S2 = blockSum(s2, red);
    const float mean = S * (1.f / DIM);
    const float inv  = rsqrtf(S2 * (1.f / DIM) - mean * mean + 1e-5f);
    for (int j = tid; j < DIM; j += 256)
        out[(size_t)i * DIM + j] = (row[j] - mean) * inv * g[j] + b[j];
}

// ---------------- final head: softmax over singleton axis == 1 ----------------
__global__ __launch_bounds__(256) void final_head(
    const float* __restrict__ x, const float* __restrict__ w,
    const float* __restrict__ b, float* __restrict__ out)
{
    const int i = blockIdx.x;
    __shared__ float red[8];
    float s = 0.f;
    for (int j = threadIdx.x; j < DIM; j += 256)
        s += x[(size_t)i * DIM + j] * w[j];
    const float z = blockSum(s, red) + b[0];
    if (threadIdx.x == 0) out[i] = __expf(z - z);  // softmax of single logit
}

// ---------------- orchestration ----------------
extern "C" void kernel_launch(void* const* d_in, const int* in_sizes, int n_in,
                              void* d_out, int out_size, void* d_ws, size_t ws_size,
                              hipStream_t stream)
{
    (void)in_sizes; (void)n_in; (void)out_size; (void)ws_size;
    const int*   inputs  = (const int*)d_in[0];
    const int*   outputs = (const int*)d_in[1];
    const float* emb_enc = (const float*)d_in[2];
    const float* emb_dec = (const float*)d_in[3];
    // param dict order: wq,bq,wk,bk,wv,bv,wfc,bfc,ln_g,ln_b,n_g,n_b
    const float* encP[12]; const float* decP[12]; const float* crsP[12];
    for (int i = 0; i < 12; ++i) {
        encP[i] = (const float*)d_in[4 + i];
        decP[i] = (const float*)d_in[16 + i];
        crsP[i] = (const float*)d_in[28 + i];
    }
    const float* fc_w = (const float*)d_in[40];
    const float* fc_b = (const float*)d_in[41];

    float* ws = (float*)d_ws;
    const size_t LD = (size_t)SEQL * DIM;    // 1M
    const size_t LH = (size_t)SEQL * HTOT;   // 8M
    float* e0  = ws;
    float* e1  = e0  + LD;
    float* dd0 = e1  + LD;
    float* dd1 = dd0 + LD;
    float* x1  = dd1 + LD;
    float* fb  = x1  + LD;
    float* qb  = fb  + LD;      // Q projection (in-place attn output)
    float* kb  = qb  + LH;
    float* vb  = kb  + LH;      // fp32 region: 120 MB

    __bf16* bws = (__bf16*)(vb + LH);
    __bf16* wtq = bws;
    __bf16* wtk = wtq + LH;
    __bf16* wtv = wtk + LH;
    __bf16* wtf = wtv + LH;
    __bf16* axA = wtf + LH;
    __bf16* axB = axA + LD;
    __bf16* axO = axB + LD;     // bf16 region: 84 MB

    auto mha = [&](const float* xq, const float* xk, const float* xv,
                   const float* const* p, float* out) {
        (void)xk; // xk == xq at all reference call sites
        dim3 gtw(HTOT / 32, DIM / 32);
        transpose_convert_bf16<<<gtw, 256, 0, stream>>>(p[0], wtq, DIM, HTOT);
        transpose_convert_bf16<<<gtw, 256, 0, stream>>>(p[2], wtk, DIM, HTOT);
        transpose_convert_bf16<<<gtw, 256, 0, stream>>>(p[4], wtv, DIM, HTOT);
        dim3 gtf(DIM / 32, HTOT / 32);
        transpose_convert_bf16<<<gtf, 256, 0, stream>>>(p[6], wtf, HTOT, DIM);
        convert_bf16<<<LD / 1024, 256, 0, stream>>>(xq, axA);
        convert_bf16<<<LD / 1024, 256, 0, stream>>>(xv, axB);
        dim3 gp(HTOT / BN, SEQL / BM);   // (64, 8)
        gemm_bias_bf16wmma<<<gp, 256, 0, stream>>>(axA, wtq, p[1], qb, DIM, HTOT);
        gemm_bias_bf16wmma<<<gp, 256, 0, stream>>>(axA, wtk, p[3], kb, DIM, HTOT);
        gemm_bias_bf16wmma<<<gp, 256, 0, stream>>>(axB, wtv, p[5], vb, DIM, HTOT);
        attn_elem<<<NHEAD * SEQL, 256, 0, stream>>>(qb, kb, vb, p[8], p[9], qb);
        convert_bf16<<<LH / 1024, 256, 0, stream>>>(qb, axO);
        dim3 gf(DIM / BN, SEQL / BM);    // (8, 8)
        gemm_bias_bf16wmma<<<gf, 256, 0, stream>>>(axO, wtf, p[7], fb, HTOT, DIM);
        add_ln<<<SEQL, 256, 0, stream>>>(fb, xq, p[10], p[11], out);
    };

    dim3 ge(SEQL, DIM / 256);
    embed_pe<<<ge, 256, 0, stream>>>(inputs,  emb_enc, e0);
    mha(e0, e0, e0, encP, e1);
    embed_pe<<<ge, 256, 0, stream>>>(outputs, emb_dec, dd0);
    mha(dd0, dd0, dd0, decP, dd1);
    mha(e1, e1, dd1, crsP, x1);
    final_head<<<SEQL, 256, 0, stream>>>(x1, fc_w, fc_b, (float*)d_out);
}
